// RNN_55207509623115
// MI455X (gfx1250) — matrix operand order repaired and checked by reading, not compiled, not run on hardware
//
#include <hip/hip_runtime.h>

typedef __bf16 bf16_t;
typedef __attribute__((ext_vector_type(16))) __bf16 v16bf;
typedef __attribute__((ext_vector_type(8)))  __bf16 v8bf;
typedef __attribute__((ext_vector_type(8)))  float  v8f;

#define OBS   32
#define LAT   256
#define HID   1024
#define BATCH 128
#define TLEN  256
#define HOR   512

// Packed-weight segment offsets in d_ws (bf16 element units).
#define OFF_WIH 0
#define OFF_WHH (OFF_WIH + LAT * OBS)
#define OFF_W1  (OFF_WHH + LAT * LAT)
#define OFF_W2  (OFF_W1  + HID * LAT)
#define OFF_W3  (OFF_W2  + HID * HID)
#define OFF_W4  (OFF_W3  + HID * HID)
#define OFF_W5  (OFF_W4  + HID * HID)
#define PACK_TOTAL (OFF_W5 + OBS * HID)

// LDS row strides (bf16 elems), +8 elems (16B) pad to spread DS banks.
#define ZS (LAT + 8)
#define HS (HID + 8)
#define IS (OBS + 8)

// ---------------------------------------------------------------------------
// Pre-pack: fp32 row-major W[N][K]  ->  bf16 WMMA B-fragment tile order.
// Tile (nt, kt) is 32 lanes x 16 bf16 contiguous (1 KB). Contiguous k order, same as load_a_frag:
//   lane L -> column n = nt*16 + L%16;  elems j=0..15 -> K = kt*32 + (L>=16?16:0) + j
// ---------------------------------------------------------------------------
__global__ void pack_w_kernel(const float* __restrict__ W, bf16_t* __restrict__ out,
                              int N, int K) {
  int total = N * K;
  int KT = K >> 5;
  for (int p = blockIdx.x * blockDim.x + threadIdx.x; p < total;
       p += gridDim.x * blockDim.x) {
    int j  = p & 15;
    int L  = (p >> 4) & 31;
    int kt = (p >> 9) % KT;
    int nt = p / (KT << 9);
    int n  = nt * 16 + (L & 15);
    int k  = kt * 32 + ((L >> 4) << 4) + j;
    out[p] = (bf16_t)W[n * K + k];
  }
}

// A-fragment (16x32 bf16) load from an LDS activation row (row = lane%16).
// elems 0..15 -> K = kb + half*16 .. +15
static __device__ __forceinline__ v16bf load_a_frag(const bf16_t* aRow, int kb, int half) {
  const v8bf a0 = *(const v8bf*)(aRow + kb + half * 16);
  const v8bf a1 = *(const v8bf*)(aRow + kb + half * 16 + 8);
  return __builtin_shufflevector(a0, a1, 0, 1, 2, 3, 4, 5, 6, 7,
                                 8, 9, 10, 11, 12, 13, 14, 15);
}

// One dense layer: out[16][N] = act( A[16][K] @ Wpacked + bias ).
// Each wave owns NT N-tiles (nt = wave + i*16); A loaded once per k-step and
// reused for all NT B streams (B offsets are compile-time-strided from one
// base, so they fold into large immediate offsets). No divergent loops, so
// no EXEC masking ever wraps a WMMA.
template <bool RELU, int NT>
static __device__ __forceinline__ void gemm_wmma(
    const bf16_t* __restrict__ aLds, int aStride, int K,
    const bf16_t* __restrict__ wPacked, const float* __restrict__ bias,
    bf16_t* __restrict__ outLds, int outStride, int wave, int lane) {
  const int m = lane & 15;
  const int half = lane >> 4;
  const int KT = K >> 5;
  const size_t ntStride = (size_t)16 * KT * 512;  // elems between a wave's tiles
  const bf16_t* aRow = aLds + m * aStride;
  const bf16_t* wpp = wPacked + (size_t)wave * KT * 512 + (size_t)lane * 16;

  v8f acc[NT];
#pragma unroll
  for (int i = 0; i < NT; ++i) {
    float bv = bias[(wave + i * 16) * 16 + m];
    v8f a = {bv, bv, bv, bv, bv, bv, bv, bv};
    acc[i] = a;
  }
#pragma unroll 4
  for (int kt = 0; kt < KT; ++kt) {
    v16bf A = load_a_frag(aRow, kt * 32, half);
#pragma unroll
    for (int i = 0; i < NT; ++i) {
      v16bf Bv = *(const v16bf*)(wpp + (size_t)kt * 512 + (size_t)i * ntStride);
      acc[i] = __builtin_amdgcn_wmma_f32_16x16x32_bf16(false, A, false, Bv,
                                                       (short)0, acc[i], false, false);
    }
  }
#pragma unroll
  for (int i = 0; i < NT; ++i) {
#pragma unroll
    for (int r = 0; r < 8; ++r) {
      float v = acc[i][r];
      if (RELU) v = fmaxf(v, 0.f);
      outLds[(half * 8 + r) * outStride + (wave + i * 16) * 16 + m] = (bf16_t)v;
    }
  }
}

// ---------------------------------------------------------------------------
// Persistent rollout: one workgroup = 16 batch rows, 16 waves, whole horizon.
// ---------------------------------------------------------------------------
__global__ __launch_bounds__(512) void rnn_rollout_kernel(
    const float* __restrict__ y, const bf16_t* __restrict__ wp,
    const float* __restrict__ b_ih, const float* __restrict__ b_hh,
    const float* __restrict__ b1, const float* __restrict__ b2,
    const float* __restrict__ b3, const float* __restrict__ b4,
    const float* __restrict__ b5, float* __restrict__ out) {
  __shared__ __align__(16) bf16_t zbuf[16 * ZS];   // latent state (bf16)
  __shared__ __align__(16) bf16_t ha[16 * HS];     // hidden ping
  __shared__ __align__(16) bf16_t hb[16 * HS];     // hidden pong
  __shared__ __align__(16) bf16_t inp[16 * IS];    // cell input (y_t or y_hat)

  const int tid = threadIdx.x;
  const int lane = tid & 31;
  const int wave = __builtin_amdgcn_readfirstlane(tid >> 5);  // SGPR-uniform
  const int m = lane & 15;
  const int half = lane >> 4;
  const int bBase = blockIdx.x * 16;

  for (int i = tid; i < 16 * ZS; i += 512) zbuf[i] = (bf16_t)0.f;
  __syncthreads();

#pragma unroll 1
  for (int t = 0; t < HOR; ++t) {
    // ---- psi(z): 5-layer MLP (each wave: 4 N-tiles per 1024-wide layer) --
    gemm_wmma<true, 4>(zbuf, ZS, LAT, wp + OFF_W1, b1, ha, HS, wave, lane);
    __syncthreads();
    gemm_wmma<true, 4>(ha, HS, HID, wp + OFF_W2, b2, hb, HS, wave, lane);
    __syncthreads();
    gemm_wmma<true, 4>(hb, HS, HID, wp + OFF_W3, b3, ha, HS, wave, lane);
    __syncthreads();
    gemm_wmma<true, 4>(ha, HS, HID, wp + OFF_W4, b4, hb, HS, wave, lane);
    __syncthreads();

    // ---- layer 5 (N=32 -> 2 tiles, waves 0..1; uniform s_cbranch) -------
    if (wave < 2) {
      const int nt = wave;
      float bv = b5[nt * 16 + m];
      v8f acc = {bv, bv, bv, bv, bv, bv, bv, bv};
      const bf16_t* aRow = hb + m * HS;
      const bf16_t* wpp =
          wp + OFF_W5 + ((size_t)nt * (HID / 32)) * 512 + (size_t)lane * 16;
#pragma unroll 4
      for (int kt = 0; kt < HID / 32; ++kt) {
        v16bf A  = load_a_frag(aRow, kt * 32, half);
        v16bf Bv = *(const v16bf*)(wpp + (size_t)kt * 512);
        acc = __builtin_amdgcn_wmma_f32_16x16x32_bf16(false, A, false, Bv,
                                                      (short)0, acc, false, false);
      }
#pragma unroll
      for (int r = 0; r < 8; ++r) {
        float v = acc[r];
        int row = half * 8 + r;
        int col = nt * 16 + m;
        out[((size_t)(bBase + row) * HOR + t) * OBS + col] = v;  // y_hat
        inp[row * IS + col] = (bf16_t)v;                         // AR input
      }
    }
    __syncthreads();

    // ---- teacher forcing: overwrite cell input with y[:, t, :] ----------
    if (t < TLEN) {
      int mm = tid >> 5, k = tid & 31;  // 512 threads cover 16x32
      inp[mm * IS + k] = (bf16_t)y[((size_t)(bBase + mm) * TLEN + t) * OBS + k];
      __syncthreads();
    }

    // ---- cell: z = tanh(inp @ W_ih^T + z @ W_hh^T + b_ih + b_hh) --------
    {
      const int nt = wave;  // 16 tiles of N=256, one per wave
      float bv = b_ih[nt * 16 + m] + b_hh[nt * 16 + m];
      v8f acc = {bv, bv, bv, bv, bv, bv, bv, bv};
      const bf16_t* aRow = zbuf + m * ZS;
      const bf16_t* wpp =
          wp + OFF_WHH + ((size_t)nt * (LAT / 32)) * 512 + (size_t)lane * 16;
#pragma unroll
      for (int kt = 0; kt < LAT / 32; ++kt) {
        v16bf A  = load_a_frag(aRow, kt * 32, half);
        v16bf Bv = *(const v16bf*)(wpp + (size_t)kt * 512);
        acc = __builtin_amdgcn_wmma_f32_16x16x32_bf16(false, A, false, Bv,
                                                      (short)0, acc, false, false);
      }
      {  // W_ih contribution: K = 32, single k-step
        v16bf A  = load_a_frag(inp + m * IS, 0, half);
        v16bf Bv = *(const v16bf*)(wp + OFF_WIH + (size_t)nt * 512 +
                                   (size_t)lane * 16);
        acc = __builtin_amdgcn_wmma_f32_16x16x32_bf16(false, A, false, Bv,
                                                      (short)0, acc, false, false);
      }
      float zn[8];
#pragma unroll
      for (int r = 0; r < 8; ++r) zn[r] = tanhf(acc[r]);
      __syncthreads();  // all reads of old z complete
#pragma unroll
      for (int r = 0; r < 8; ++r)
        zbuf[(half * 8 + r) * ZS + nt * 16 + m] = (bf16_t)zn[r];
      __syncthreads();
    }
  }
}

extern "C" void kernel_launch(void* const* d_in, const int* in_sizes, int n_in,
                              void* d_out, int out_size, void* d_ws, size_t ws_size,
                              hipStream_t stream) {
  const float* y    = (const float*)d_in[0];
  const float* W_ih = (const float*)d_in[2];
  const float* W_hh = (const float*)d_in[3];
  const float* b_ih = (const float*)d_in[4];
  const float* b_hh = (const float*)d_in[5];
  const float* W1 = (const float*)d_in[6];  const float* b1 = (const float*)d_in[7];
  const float* W2 = (const float*)d_in[8];  const float* b2 = (const float*)d_in[9];
  const float* W3 = (const float*)d_in[10]; const float* b3 = (const float*)d_in[11];
  const float* W4 = (const float*)d_in[12]; const float* b4 = (const float*)d_in[13];
  const float* W5 = (const float*)d_in[14]; const float* b5 = (const float*)d_in[15];
  bf16_t* wp = (bf16_t*)d_ws;

  auto pack = [&](const float* W, bf16_t* o, int N, int K) {
    int total = N * K;
    pack_w_kernel<<<(total + 255) / 256, 256, 0, stream>>>(W, o, N, K);
  };
  pack(W_ih, wp + OFF_WIH, LAT, OBS);
  pack(W_hh, wp + OFF_WHH, LAT, LAT);
  pack(W1,   wp + OFF_W1,  HID, LAT);
  pack(W2,   wp + OFF_W2,  HID, HID);
  pack(W3,   wp + OFF_W3,  HID, HID);
  pack(W4,   wp + OFF_W4,  OBS * 0 + HID, HID);
  pack(W5,   wp + OFF_W5,  OBS, HID);

  rnn_rollout_kernel<<<BATCH / 16, 512, 0, stream>>>(
      y, wp, b_ih, b_hh, b1, b2, b3, b4, b5, (float*)d_out);
}
